// MultiHeadAttention_41583873360177
// MI455X (gfx1250) — compile-verified, matrix operand order checked by reading, not run on hardware
//
#include <hip/hip_runtime.h>
#include <hip/hip_bf16.h>
#include <math.h>

// ---------------------------------------------------------------------------
// MHA forward for MI455X (gfx1250): bf16 WMMA GEMMs + flash attention.
// Double-buffered TDM (tensor_load_to_lds, TENSORcnt<=2) pipelines the GEMM
// tile DMA under the WMMAs; attention K tiles double-buffered through
// global_load_async_to_lds_b128 (ASYNCcnt<=4).
// B=2, S=2048, D_MODEL=1024, HEADS=16, D_K=64.
// ---------------------------------------------------------------------------

#define BATCH   2
#define SEQ     2048
#define DM      1024
#define HEADS   16
#define DK      64
#define TOK     (BATCH * SEQ)   // 4096

typedef __attribute__((ext_vector_type(16))) __bf16 bf16x16;
typedef __attribute__((ext_vector_type(8)))  __bf16 bf16x8;
typedef __attribute__((ext_vector_type(4)))  __bf16 bf16x4;
typedef __attribute__((ext_vector_type(8)))  float  f32x8;
typedef __attribute__((ext_vector_type(4)))  unsigned int u32x4;
typedef __attribute__((ext_vector_type(8)))  int    i32x8;
typedef __attribute__((ext_vector_type(4)))  int    i32x4;

// Build a 16x32 A/B fragment half for this lane from two contiguous 8-elem
// (16-byte) runs, per the CDNA5 16-bit matrix VGPR layout (ISA 7.12.2):
//   lanes 0-15 : K = {klo..klo+7} u {klo+16..klo+23} with klo=0
//   lanes 16-31: same with klo=8
__device__ __forceinline__ bf16x16 make_frag(const __bf16* p0) {
    bf16x8 lo = *(const bf16x8*)(p0);
    bf16x8 hi = *(const bf16x8*)(p0 + 16);
    return __builtin_shufflevector(lo, hi, 0,1,2,3,4,5,6,7,8,9,10,11,12,13,14,15);
}

__device__ __forceinline__ f32x8 wmma_bf16(bf16x16 a, bf16x16 b, f32x8 c) {
    return __builtin_amdgcn_wmma_f32_16x16x32_bf16(
        /*neg_a=*/false, a, /*neg_b=*/false, b,
        /*c_mod=*/(short)0, c, /*reuse_a=*/false, /*reuse_b=*/false);
}

// Low 32 bits of a generic pointer to LDS == byte offset in LDS aperture
// (ISA 10.2).
__device__ __forceinline__ unsigned lds_off_u32(const void* p) {
    return (unsigned)(unsigned long long)p;
}

__device__ __forceinline__ void wait_tensorcnt0() {
#if __has_builtin(__builtin_amdgcn_s_wait_tensorcnt)
    __builtin_amdgcn_s_wait_tensorcnt((short)0);
#else
    asm volatile("s_wait_tensorcnt 0x0" ::: "memory");
#endif
}
__device__ __forceinline__ void wait_tensorcnt2() {
#if __has_builtin(__builtin_amdgcn_s_wait_tensorcnt)
    __builtin_amdgcn_s_wait_tensorcnt((short)2);
#else
    asm volatile("s_wait_tensorcnt 0x2" ::: "memory");
#endif
}
__device__ __forceinline__ void wait_asynccnt0() {
#if __has_builtin(__builtin_amdgcn_s_wait_asynccnt)
    __builtin_amdgcn_s_wait_asynccnt((short)0);
#else
    asm volatile("s_wait_asynccnt 0x0" ::: "memory");
#endif
}
__device__ __forceinline__ void wait_asynccnt4() {
#if __has_builtin(__builtin_amdgcn_s_wait_asynccnt)
    __builtin_amdgcn_s_wait_asynccnt((short)4);
#else
    asm volatile("s_wait_asynccnt 0x4" ::: "memory");
#endif
}

// ---------------------------------------------------------------------------
// Kernel 1: f32 -> bf16 pack (vectorized)
// ---------------------------------------------------------------------------
__global__ void f32_to_bf16(const float* __restrict__ in,
                            __bf16* __restrict__ out, int n) {
    int i = (blockIdx.x * blockDim.x + threadIdx.x) * 4;
    if (i + 4 <= n) {
        float4 f = *(const float4*)(in + i);
        bf16x4 o = { (__bf16)f.x, (__bf16)f.y, (__bf16)f.z, (__bf16)f.w };
        *(bf16x4*)(out + i) = o;
    } else {
        for (; i < n; ++i) out[i] = (__bf16)in[i];
    }
}

// ---------------------------------------------------------------------------
// Kernel 2: Y = X @ W^T + bias   (X:[M,K] bf16, W:[N,K] bf16 row-major)
// Block: 256 thr (8 waves), 128x128 output tile.  K tiles (BK=32) streamed
// into double-buffered LDS by the Tensor Data Mover; at tile t wave 0 issues
// the t+1 DMA then waits TENSORcnt<=2 so the DMA overlaps the 64 WMMAs.
// mode 0: write bf16 head-split [B,H,S,DK];  mode 1: write f32 flat [M,N].
// ---------------------------------------------------------------------------
#define BM  128
#define BN  128
#define BK  32
#define LDA 40   // LDS row stride in bf16: 16 DWORDs data + 4 DWORDs TDM pad

#if __has_builtin(__builtin_amdgcn_tensor_load_to_lds)
#define HAVE_TDM 1
// TDM load of a [rows x BK] bf16 tile whose rows are `rowlen` elements apart,
// into LDS at `lds`, padded to LDA stride (pad_interval=16 DW, pad_amount=4 DW).
// D# layout per CDNA5 ISA 8.3/8.4.  All fields wave-uniform -> SGPR groups.
__device__ __forceinline__ void tdm_load_tile(unsigned lds, const __bf16* gptr,
                                              unsigned rows, unsigned rowlen) {
    unsigned long long ga = (unsigned long long)(size_t)gptr;
    u32x4 g0;
    g0[0] = 1u;                                          // count=1 (user D#)
    g0[1] = lds;                                         // lds_addr
    g0[2] = (unsigned)(ga & 0xffffffffu);                // global_addr[31:0]
    g0[3] = (unsigned)((ga >> 32) & 0x1ffffffu)          // global_addr[56:32]
          | (2u << 30);                                  // type = 2 (image)
    i32x8 g1;
    g1[0] = (int)((1u << 16)      // data_size = 1 (2 bytes)
          |       (1u << 20)      // pad_enable
          |       (3u << 22)      // pad_interval: 16 DWORDs between pads
          |       (3u << 25));    // pad_amount:   4 DWORDs of pad
    g1[1] = (int)((rowlen & 0xffffu) << 16);             // tensor_dim0[15:0]
    g1[2] = (int)(((rowlen >> 16) & 0xffffu)             // tensor_dim0[31:16]
          |       ((rows & 0xffffu) << 16));             // tensor_dim1[15:0]
    g1[3] = (int)(((rows >> 16) & 0xffffu)               // tensor_dim1[31:16]
          |       ((unsigned)BK << 16));                 // tile_dim0 = 32
    g1[4] = (int)rows;                                   // tile_dim1; tile_dim2=0
    g1[5] = (int)rowlen;                                 // tensor_dim0_stride lo32
    g1[6] = 0;                                           // stride hi / dim1_stride
    g1[7] = 0;
    i32x4 g2 = {0, 0, 0, 0};
    i32x4 g3 = {0, 0, 0, 0};
#if __clang_major__ >= 23
    i32x8 g4 = {};
    __builtin_amdgcn_tensor_load_to_lds(g0, g1, g2, g3, g4, 0);
#else
    __builtin_amdgcn_tensor_load_to_lds(g0, g1, g2, g3, 0);
#endif
}
#else
#define HAVE_TDM 0
#endif

__global__ void __launch_bounds__(256)
gemm_xwt(const __bf16* __restrict__ X, const __bf16* __restrict__ W,
         const float* __restrict__ bias,
         __bf16* __restrict__ outHead, float* __restrict__ outFlat,
         int M, int N, int K, int mode)
{
    __shared__ __align__(16) __bf16 As[2][BM * LDA];
    __shared__ __align__(16) __bf16 Bs[2][BN * LDA];

    const int tid  = threadIdx.x;
    const int lane = tid & 31;
    const int wave = tid >> 5;
    const int wm   = (wave & 3) * 32;   // wave row offset in block tile
    const int wn   = (wave >> 2) * 64;  // wave col offset in block tile
    const int bm   = blockIdx.x * BM;
    const int bn   = blockIdx.y * BN;

    const int frow = lane & 15;                 // fragment row (A) / col (B)
    const int klo  = (lane & 16) ? 8 : 0;       // K half offset per lane group

    f32x8 acc[2][4] = {};
    const int NT = K / BK;

#if HAVE_TDM
    if (wave == 0) {   // prologue: DMA tile 0; EXEC ignored by TDM
        tdm_load_tile(lds_off_u32(As[0]), &X[(size_t)bm * K],
                      (unsigned)BM, (unsigned)K);
        tdm_load_tile(lds_off_u32(Bs[0]), &W[(size_t)bn * K],
                      (unsigned)BN, (unsigned)K);
    }
#endif

    for (int t = 0; t < NT; ++t) {
        const int cur = t & 1;
#if HAVE_TDM
        if (wave == 0) {
            if (t + 1 < NT) {   // DMA next tile into the other buffer, then
                                // wait for the 2 oldest descriptors (tile t)
                tdm_load_tile(lds_off_u32(As[cur ^ 1]),
                              &X[(size_t)bm * K + (t + 1) * BK],
                              (unsigned)BM, (unsigned)K);
                tdm_load_tile(lds_off_u32(Bs[cur ^ 1]),
                              &W[(size_t)bn * K + (t + 1) * BK],
                              (unsigned)BN, (unsigned)K);
                wait_tensorcnt2();
            } else {
                wait_tensorcnt0();
            }
        }
#else
        #pragma unroll
        for (int c = 0; c < 2; ++c) {
            int chunk = tid + c * 256;          // 512 chunks of 8 bf16
            int row   = chunk >> 2;
            int col   = (chunk & 3) * 8;
            *(bf16x8*)&As[cur][row * LDA + col] =
                *(const bf16x8*)&X[(size_t)(bm + row) * K + t * BK + col];
            *(bf16x8*)&Bs[cur][row * LDA + col] =
                *(const bf16x8*)&W[(size_t)(bn + row) * K + t * BK + col];
        }
#endif
        __syncthreads();   // tile t visible to all waves

        if (t + 2 < NT) {  // hint tile t+2 toward L2 (global_prefetch_b8)
            int row = tid >> 2, col = (tid & 3) * 8;
            __builtin_prefetch(&X[(size_t)(bm + row) * K + (t + 2) * BK + col], 0, 1);
        }

        #pragma unroll
        for (int mt = 0; mt < 2; ++mt) {
            bf16x16 a = make_frag(&As[cur][(wm + mt * 16 + frow) * LDA + klo]);
            #pragma unroll
            for (int nt = 0; nt < 4; ++nt) {
                bf16x16 b = make_frag(&Bs[cur][(wn + nt * 16 + frow) * LDA + klo]);
                acc[mt][nt] = wmma_bf16(a, b, acc[mt][nt]);
            }
        }
        __syncthreads();   // reads of buffer `cur` done before its next DMA
    }

    // Epilogue: C layout -> lane col = lane&15, VGPR r -> row r + 8*(lane>>4)
    const int rhalf = (lane >> 4) * 8;
    #pragma unroll
    for (int mt = 0; mt < 2; ++mt) {
        #pragma unroll
        for (int nt = 0; nt < 4; ++nt) {
            #pragma unroll
            for (int r = 0; r < 8; ++r) {
                int m = bm + wm + mt * 16 + rhalf + r;
                int n = bn + wn + nt * 16 + (lane & 15);
                float v = acc[mt][nt][r] + bias[n];
                if (mode == 0) {
                    int bb = m >> 11, s = m & (SEQ - 1);
                    int h  = n >> 6,  d = n & (DK - 1);
                    outHead[(((size_t)(bb * HEADS + h)) * SEQ + s) * DK + d] = (__bf16)v;
                } else {
                    outFlat[(size_t)m * N + n] = v;
                }
            }
        }
    }
}

// ---------------------------------------------------------------------------
// Kernel 3: flash attention.  Grid (S/64, HEADS, BATCH), 128 thr (4 waves).
// Each wave owns 16 query rows; loop over 32 key blocks of 64.
// K tiles double-buffered via global_load_async_to_lds_b128: block j+1's
// copies are issued before waiting ASYNCcnt<=4 (oldest 4 = block j, async
// loads complete in order), so the copy overlaps softmax + both WMMAs.
// ---------------------------------------------------------------------------
#define PST 72   // LDS row stride (144B, 16B aligned, bank-skewed)

__global__ void __launch_bounds__(128)
attn_flash(const __bf16* __restrict__ Qh, const __bf16* __restrict__ Kh,
           const __bf16* __restrict__ Vh, __bf16* __restrict__ Oc)
{
    __shared__ __align__(16) __bf16 Ks[2][64 * PST]; // K block [n][d], 2 bufs
    __shared__ __align__(16) __bf16 Vt[64 * PST];    // V block transposed [d][n]
    __shared__ __align__(16) __bf16 Ps[64 * PST];    // probabilities [m][n]

    const int tid  = threadIdx.x;
    const int lane = tid & 31;
    const int wave = tid >> 5;
    const int qb   = blockIdx.x;
    const int h    = blockIdx.y;
    const int b    = blockIdx.z;

    const size_t headBase = (size_t)(b * HEADS + h) * SEQ * DK;
    const int frow  = lane & 15;
    const int klo   = (lane & 16) ? 8 : 0;
    const int rhalf = (lane >> 4) * 8;

    // This thread's 4 staging chunks (shared by K async copy and V loads).
    int crow[4], ccol[4];
    #pragma unroll
    for (int c = 0; c < 4; ++c) {
        int chunk = tid + c * 128;       // 512 chunks of 8 bf16
        crow[c] = chunk >> 3;
        ccol[c] = (chunk & 7) * 8;
    }

    auto issue_k_async = [&](int blk, int buf) {
        #pragma unroll
        for (int c = 0; c < 4; ++c) {
            unsigned    ldsK = lds_off_u32(&Ks[buf][crow[c] * PST + ccol[c]]);
            const __bf16* gk =
                &Kh[headBase + (size_t)(blk * 64 + crow[c]) * DK + ccol[c]];
            asm volatile("global_load_async_to_lds_b128 %0, %1, off"
                         :: "v"(ldsK), "v"(gk) : "memory");
        }
    };

    // Q fragments for this wave's 16 rows (A layout reads contiguous in d).
    const int qrow = qb * 64 + wave * 16 + frow;
    bf16x16 qf[2];
    #pragma unroll
    for (int kt = 0; kt < 2; ++kt)
        qf[kt] = make_frag(Qh + headBase + (size_t)qrow * DK + kt * 32 + klo);

    f32x8 ot[4] = {};
    float rowM[8], rowL[8];
    #pragma unroll
    for (int r = 0; r < 8; ++r) { rowM[r] = -3.0e38f; rowL[r] = 0.f; }

    issue_k_async(0, 0);   // prologue: start block 0's K copy

    for (int j = 0; j < 32; ++j) {
        const int cur = j & 1;
        // Stage V for block j via VGPRs with transpose (it must visit VGPRs
        // anyway); meanwhile kick off block j+1's K async copy, then retire
        // block j's (oldest 4 async ops).
        #pragma unroll
        for (int c = 0; c < 4; ++c) {
            bf16x8 vv = *(const bf16x8*)
                &Vh[headBase + (size_t)(j * 64 + crow[c]) * DK + ccol[c]];
            #pragma unroll
            for (int e = 0; e < 8; ++e) Vt[(ccol[c] + e) * PST + crow[c]] = vv[e];
        }
        if (j + 1 < 32) {
            issue_k_async(j + 1, cur ^ 1);
            wait_asynccnt4();
        } else {
            wait_asynccnt0();
        }
        __syncthreads();

        // S = (Q K^T) / sqrt(64): 4 col tiles x 2 K-steps of WMMA.
        f32x8 st[4] = {};
        #pragma unroll
        for (int kt = 0; kt < 2; ++kt) {
            #pragma unroll
            for (int nt = 0; nt < 4; ++nt) {
                bf16x16 kf =
                    make_frag(&Ks[cur][(nt * 16 + frow) * PST + kt * 32 + klo]);
                st[nt] = wmma_bf16(qf[kt], kf, st[nt]);
            }
        }

        // Online softmax per row (row r lives in one 16-lane half-group).
        #pragma unroll
        for (int r = 0; r < 8; ++r) {
            float mx = rowM[r];
            #pragma unroll
            for (int nt = 0; nt < 4; ++nt) {
                st[nt][r] *= 0.125f;         // 1/sqrt(DK)
                mx = fmaxf(mx, st[nt][r]);
            }
            #pragma unroll
            for (int off = 1; off < 16; off <<= 1)
                mx = fmaxf(mx, __shfl_xor(mx, off, 32));
            float alpha = __expf(rowM[r] - mx);
            rowM[r] = mx;
            float sum = 0.f;
            #pragma unroll
            for (int nt = 0; nt < 4; ++nt) {
                float pv = __expf(st[nt][r] - mx);
                st[nt][r] = pv;
                sum += pv;
            }
            #pragma unroll
            for (int off = 1; off < 16; off <<= 1)
                sum += __shfl_xor(sum, off, 32);
            rowL[r] = rowL[r] * alpha + sum;
            #pragma unroll
            for (int dt = 0; dt < 4; ++dt) ot[dt][r] *= alpha;
        }

        // C-layout -> A-layout via LDS: write P (bf16) by [m][n].
        #pragma unroll
        for (int nt = 0; nt < 4; ++nt)
            #pragma unroll
            for (int r = 0; r < 8; ++r)
                Ps[(wave * 16 + rhalf + r) * PST + nt * 16 + (lane & 15)] =
                    (__bf16)st[nt][r];
        __syncthreads();

        // O += P @ V (V transposed in LDS so B-frags load contiguously).
        #pragma unroll
        for (int kt = 0; kt < 2; ++kt) {
            bf16x16 pf = make_frag(&Ps[(wave * 16 + frow) * PST + kt * 32 + klo]);
            #pragma unroll
            for (int dt = 0; dt < 4; ++dt) {
                bf16x16 vf = make_frag(&Vt[(dt * 16 + frow) * PST + kt * 32 + klo]);
                ot[dt] = wmma_bf16(pf, vf, ot[dt]);
            }
        }
        __syncthreads();   // Vt/Ps reads done before next iteration rewrites
    }

    // Normalize and write concat layout [B, S, H*DK] as bf16 for final GEMM.
    #pragma unroll
    for (int dt = 0; dt < 4; ++dt) {
        #pragma unroll
        for (int r = 0; r < 8; ++r) {
            int m   = qb * 64 + wave * 16 + rhalf + r;
            int col = h * DK + dt * 16 + (lane & 15);
            float v = ot[dt][r] / rowL[r];
            Oc[((size_t)(b * SEQ + m)) * DM + col] = (__bf16)v;
        }
    }
}

// ---------------------------------------------------------------------------
// Launch
// ---------------------------------------------------------------------------
extern "C" void kernel_launch(void* const* d_in, const int* in_sizes, int n_in,
                              void* d_out, int out_size, void* d_ws, size_t ws_size,
                              hipStream_t stream) {
    const float* q  = (const float*)d_in[0];
    const float* k  = (const float*)d_in[1];
    const float* v  = (const float*)d_in[2];
    const float* Wq = (const float*)d_in[3];
    const float* bq = (const float*)d_in[4];
    const float* Wk = (const float*)d_in[5];
    const float* bk = (const float*)d_in[6];
    const float* Wv = (const float*)d_in[7];
    const float* bv = (const float*)d_in[8];
    const float* Wo = (const float*)d_in[9];
    const float* bo = (const float*)d_in[10];

    const size_t XN = (size_t)TOK * DM;   // 4M elems
    const size_t WN = (size_t)DM * DM;    // 1M elems

    __bf16* ws = (__bf16*)d_ws;
    __bf16* xq = ws;             // bf16 copies of activations
    __bf16* xk = xq + XN;
    __bf16* xv = xk + XN;
    __bf16* wq = xv + XN;        // bf16 weights
    __bf16* wk = wq + WN;
    __bf16* wv = wk + WN;
    __bf16* wo = wv + WN;
    __bf16* qh = wo + WN;        // head-split projections [B,H,S,DK]
    __bf16* kh = qh + XN;
    __bf16* vh = kh + XN;
    __bf16* ao = vh + XN;        // attention output concat [TOK, DM]

    auto conv = [&](const float* in, __bf16* out, size_t n) {
        int nb = (int)((n / 4 + 255) / 256);
        f32_to_bf16<<<nb, 256, 0, stream>>>(in, out, (int)n);
    };
    conv(q,  xq, XN);
    conv(k,  xk, XN);
    conv(v,  xv, XN);
    conv(Wq, wq, WN);
    conv(Wk, wk, WN);
    conv(Wv, wv, WN);
    conv(Wo, wo, WN);

    dim3 gg(TOK / BM, DM / BN);
    gemm_xwt<<<gg, 256, 0, stream>>>(xq, wq, bq, qh, nullptr, TOK, DM, DM, 0);
    gemm_xwt<<<gg, 256, 0, stream>>>(xk, wk, bk, kh, nullptr, TOK, DM, DM, 0);
    gemm_xwt<<<gg, 256, 0, stream>>>(xv, wv, bv, vh, nullptr, TOK, DM, DM, 0);

    attn_flash<<<dim3(SEQ / 64, HEADS, BATCH), 128, 0, stream>>>(qh, kh, vh, ao);

    gemm_xwt<<<gg, 256, 0, stream>>>(ao, wo, bo, nullptr, (float*)d_out,
                                     TOK, DM, DM, 1);
}
